// WindowedAttention_18683107737836
// MI455X (gfx1250) — matrix operand order repaired and checked by reading, not compiled, not run on hardware
//
#include <hip/hip_runtime.h>

#define DIMD 1024
#define KW 7
#define SEQN 4096
#define SEQM 1024
#define NHEADS 16
#define HD 64
#define BATCH 2
#define EPS 1e-5f
#define CSTEPS (KW * (DIMD / 32))   // 224 K-steps of 32
#define NCHUNKS (SEQN / 32)         // 128 key chunks

typedef __attribute__((ext_vector_type(16))) __bf16 v16bf;
typedef __attribute__((ext_vector_type(8)))  float  v8f;

union Frag {
    v16bf v;
    uint4 u[2];
    unsigned short s[16];
};

__device__ __forceinline__ unsigned short f2bf(float f) {
    union { float f; unsigned u; } v; v.f = f;
    unsigned r = v.u + 0x7FFFu + ((v.u >> 16) & 1u);
    return (unsigned short)(r >> 16);
}

template <int MASK>
__device__ __forceinline__ float swz(float x) {
    union { float f; int i; } u; u.f = x;
    // group-of-32 swizzle: and_mask=0x1F, or_mask=0, xor_mask=MASK
    u.i = __builtin_amdgcn_ds_swizzle(u.i, 0x1F | (MASK << 10));
    return u.f;
}

// ASYNCcnt-tracked global->LDS 16B copy (no VGPR round trip). GVS mode:
// mem = SADDR(base) + VADDR(voff32); LDS dest = VDST(lds addr).
__device__ __forceinline__ void async_b128(unsigned lds_addr, unsigned voff,
                                           const void* sbase) {
    asm volatile("global_load_async_to_lds_b128 %0, %1, %2"
                 :: "v"(lds_addr), "v"(voff), "s"(sbase) : "memory");
}
__device__ __forceinline__ void wait_async0() {
    asm volatile("s_wait_asynccnt 0x0" ::: "memory");
}
__device__ __forceinline__ unsigned lds_addr_of(const void* p) {
    return (unsigned)(size_t)p;   // low 32 bits of generic addr == LDS offset
}

// ---------------------------------------------------------------------------
// 1) LayerNorm: x (f32, rows of 1024) -> xn (bf16)
// ---------------------------------------------------------------------------
__global__ __launch_bounds__(256)
void ln_kernel(const float* __restrict__ x, const float* __restrict__ w,
               const float* __restrict__ b, unsigned short* __restrict__ xn) {
    const int row = blockIdx.x;
    const int tid = threadIdx.x;
    const size_t base = (size_t)row * DIMD + tid * 4;

    float4 v = *(const float4*)(x + base);
    float s  = v.x + v.y + v.z + v.w;
    float sq = v.x * v.x + v.y * v.y + v.z * v.z + v.w * v.w;

    s  += swz<1>(s);  s  += swz<2>(s);  s  += swz<4>(s);  s  += swz<8>(s);  s  += swz<16>(s);
    sq += swz<1>(sq); sq += swz<2>(sq); sq += swz<4>(sq); sq += swz<8>(sq); sq += swz<16>(sq);

    __shared__ float redS[8], redQ[8];
    const int wave = tid >> 5, lane = tid & 31;
    if (lane == 0) { redS[wave] = s; redQ[wave] = sq; }
    __syncthreads();
    float ts = 0.f, tq = 0.f;
#pragma unroll
    for (int i = 0; i < 8; ++i) { ts += redS[i]; tq += redQ[i]; }

    const float mu   = ts * (1.0f / DIMD);
    const float var  = tq * (1.0f / DIMD) - mu * mu;
    const float rstd = rsqrtf(var + EPS);

    const int c = tid * 4;
    xn[base + 0] = f2bf((v.x - mu) * rstd * w[c + 0] + b[c + 0]);
    xn[base + 1] = f2bf((v.y - mu) * rstd * w[c + 1] + b[c + 1]);
    xn[base + 2] = f2bf((v.z - mu) * rstd * w[c + 2] + b[c + 2]);
    xn[base + 3] = f2bf((v.w - mu) * rstd * w[c + 3] + b[c + 3]);
}

// ---------------------------------------------------------------------------
// 2) Weight repack: W[o][i][k] f32 -> Wt2[k][o][i] bf16  (B-tiles become
//    linear in memory -> async-copyable straight into LDS, no transpose)
// ---------------------------------------------------------------------------
__global__ __launch_bounds__(256)
void wcvt_kernel(const float* __restrict__ W, unsigned short* __restrict__ Wt2) {
    const int idx = blockIdx.x * 256 + threadIdx.x;
    const int k   = idx / (DIMD * DIMD);
    const int rem = idx % (DIMD * DIMD);
    const int o   = rem / DIMD;
    const int i   = rem % DIMD;
    Wt2[idx] = f2bf(W[((size_t)o * DIMD + i) * KW + k]);
}

// ---------------------------------------------------------------------------
// 3) Conv1d(K=7) as tap-major GEMM, bf16 WMMA. Double-buffered LDS; BOTH
//    activation and weight tiles staged by global_load_async_to_lds_b128.
//    Block = 256 thr (8 waves); block tile = 256 rows x 64 out-ch;
//    wave tile = 32 x 64 -> 8 WMMAs / K-step. B-frags preloaded into
//    distinct regs so dscnt waits are partial, not full drains.
// ---------------------------------------------------------------------------
__global__ __launch_bounds__(256)
void conv_gemm_kernel(const unsigned short* __restrict__ xn,    // bf16 (B,N,D)
                      const unsigned short* __restrict__ Wt2,   // bf16 [7][o][i]
                      unsigned short* __restrict__ outb,        // bf16 rows x D | null
                      float* __restrict__ outf,                 // f32 rows x D | null
                      unsigned short* __restrict__ outT,        // bf16 [b*D+ch][N] | null
                      int rowsPerBatch, int seqStride) {
    __shared__ __align__(16) unsigned short Alds[2][256][32];   // rows x k
    __shared__ __align__(16) unsigned short Blds[2][64][32];    // col  x k

    const int tid  = threadIdx.x;
    const int wave = tid >> 5, lane = tid & 31;
    const int row16 = lane & 15, half = lane >> 4;
    const int rowBase = blockIdx.x * 256;
    const int colBase = blockIdx.y * 64;
    const int b = rowBase / rowsPerBatch;                       // uniform per block
    const unsigned short* xb = xn + (size_t)b * SEQN * DIMD;    // uniform SGPR base

    v8f acc[2][4];
#pragma unroll
    for (int rt = 0; rt < 2; ++rt)
#pragma unroll
        for (int t = 0; t < 4; ++t) acc[rt][t] = v8f{};

    // ---- async stage A tile (256 rows x 32 ch) for step s ----
    auto stageA = [&](int s, int buf) {
        const int tap = s >> 5;
        const int c0  = (s & 31) << 5;
#pragma unroll
        for (int p = 0; p < 4; ++p) {
            const int chunk = tid + p * 256;          // 0..1023
            const int r   = chunk >> 2;
            const int off = (chunk & 3) * 8;
            const int n   = (rowBase + r) % rowsPerBatch;
            const int seq = n * seqStride + tap - (KW / 2);
            if (seq >= 0 && seq < SEQN) {
                const unsigned voff = (unsigned)((seq * DIMD + c0 + off) * 2);
                async_b128(lds_addr_of(&Alds[buf][r][off]), voff, xb);
            } else {
                *(uint4*)(&Alds[buf][r][off]) = make_uint4(0u, 0u, 0u, 0u);
            }
        }
    };
    // ---- async stage B tile (64 cols x 32 k) for step s (linear copy) ----
    auto stageB = [&](int s, int buf) {
        const int tap = s >> 5;
        const int c0  = (s & 31) << 5;
        const int colrow = tid >> 2;                  // 0..63
        const int koff   = (tid & 3) * 8;
        const unsigned voff =
            (unsigned)((((tap * DIMD + colBase + colrow) * DIMD) + c0 + koff) * 2);
        async_b128(lds_addr_of(&Blds[buf][colrow][koff]), voff, Wt2);
    };

    stageA(0, 0);
    stageB(0, 0);
    wait_async0();
    __syncthreads();

    for (int s = 0; s < CSTEPS; ++s) {
        const int cur = s & 1, nxt = cur ^ 1;
        if (s + 1 < CSTEPS) { stageA(s + 1, nxt); stageB(s + 1, nxt); }
        // ---- preload ALL fragments for this step, then the WMMA chain ----
        const int rb = wave * 32;
        Frag a0, a1, bf[4];
        a0.u[0] = *(const uint4*)(&Alds[cur][rb + row16][half * 8]);
        a0.u[1] = *(const uint4*)(&Alds[cur][rb + row16][16 + half * 8]);
        a1.u[0] = *(const uint4*)(&Alds[cur][rb + 16 + row16][half * 8]);
        a1.u[1] = *(const uint4*)(&Alds[cur][rb + 16 + row16][16 + half * 8]);
#pragma unroll
        for (int t = 0; t < 4; ++t) {
            bf[t].u[0] = *(const uint4*)(&Blds[cur][t * 16 + row16][half * 8]);
            bf[t].u[1] = *(const uint4*)(&Blds[cur][t * 16 + row16][16 + half * 8]);
        }
#pragma unroll
        for (int t = 0; t < 4; ++t) {
            acc[0][t] = __builtin_amdgcn_wmma_f32_16x16x32_bf16(
                false, a0.v, false, bf[t].v, (short)0, acc[0][t], false, false);
            acc[1][t] = __builtin_amdgcn_wmma_f32_16x16x32_bf16(
                false, a1.v, false, bf[t].v, (short)0, acc[1][t], false, false);
        }
        wait_async0();
        __syncthreads();
    }

    // ---- epilogue ----
    const int col = lane & 15;
    const int nBase = (rowBase % rowsPerBatch) + wave * 32;
#pragma unroll
    for (int rt = 0; rt < 2; ++rt) {
#pragma unroll
        for (int t = 0; t < 4; ++t) {
            if (outb) {
#pragma unroll
                for (int r = 0; r < 8; ++r) {
                    const int m = rowBase + wave * 32 + rt * 16 + 8 * half + r;
                    const int o = colBase + t * 16 + col;
                    const float val = acc[rt][t][r];
                    outb[(size_t)m * DIMD + o] = f2bf(val);
                    if (outf) outf[(size_t)m * DIMD + o] = val;
                }
            }
            if (outT) {   // lane's 8 rows are consecutive n -> one packed b128
                unsigned short pk[8];
#pragma unroll
                for (int r = 0; r < 8; ++r) pk[r] = f2bf(acc[rt][t][r]);
                const int ch = colBase + t * 16 + col;
                const int n  = nBase + rt * 16 + 8 * half;
                *(uint4*)(outT + ((size_t)(b * DIMD + ch)) * SEQN + n) = *(const uint4*)pk;
            }
        }
    }
}

// ---------------------------------------------------------------------------
// 4) Windowed attention, flash style, block-cooperative over keys.
//    Block = 8 waves, each owns 16 query rows of one (b,h); all waves walk
//    the key dim in lockstep sharing double-buffered async-staged K / Vt.
// ---------------------------------------------------------------------------
__global__ __launch_bounds__(256)
void attn_kernel(const unsigned short* __restrict__ Qbf,
                 const unsigned short* __restrict__ Kbf,   // row-major (b,n,D)
                 const unsigned short* __restrict__ Vt,    // transposed [b*D+ch][N]
                 const float* __restrict__ Qf, float* __restrict__ out) {
    __shared__ __align__(16) unsigned short Klds[2][32][64];  // key x hd
    __shared__ __align__(16) unsigned short Vlds[2][64][32];  // ch  x key
    __shared__ __align__(16) unsigned short Slds[8][16][32];  // per-wave P tile

    const int tid  = threadIdx.x;
    const int wave = tid >> 5, lane = tid & 31;
    const int col  = lane & 15, half = lane >> 4, row = lane & 15;
    const int bh = blockIdx.y;
    const int bb = bh >> 4, h = bh & 15;
    const int mBase = blockIdx.x * 128 + wave * 16;

    const unsigned short* Qb = Qbf + ((size_t)bb * SEQM + mBase) * DIMD + h * HD;
    const unsigned short* Kb = Kbf + (size_t)bb * SEQN * DIMD + h * HD;      // uniform
    const unsigned short* Vb = Vt + ((size_t)(bb * DIMD + h * HD)) * SEQN;   // uniform

    // K stage: 32 keys x 64 hd (128B rows) : thread -> (key, 8ch)
    const int kKey = tid >> 3, kOff = (tid & 7) * 8;
    // V stage: 64 ch x 32 keys (64B rows)  : thread -> (ch, 8keys)
    const int vCh = tid >> 2, vOff = (tid & 3) * 8;

    auto stageKV = [&](int n0, int buf) {
        async_b128(lds_addr_of(&Klds[buf][kKey][kOff]),
                   (unsigned)(((n0 + kKey) * DIMD + kOff) * 2), Kb);
        async_b128(lds_addr_of(&Vlds[buf][vCh][vOff]),
                   (unsigned)((vCh * SEQN + n0 + vOff) * 2), Vb);
    };

    // Q fragments (A layout 16x32), two k-chunks covering hd=64
    Frag qf[2];
#pragma unroll
    for (int kc = 0; kc < 2; ++kc) {
        qf[kc].u[0] = *(const uint4*)(Qb + (size_t)row * DIMD + kc * 32 + half * 8);
        qf[kc].u[1] = *(const uint4*)(Qb + (size_t)row * DIMD + kc * 32 + 16 + half * 8);
    }

    v8f o[4] = {v8f{}, v8f{}, v8f{}, v8f{}};
    float mrow[8], lrow[8];
#pragma unroll
    for (int r = 0; r < 8; ++r) { mrow[r] = -3.0e38f; lrow[r] = 0.f; }

    stageKV(0, 0);
    wait_async0();
    __syncthreads();

    for (int it = 0; it < NCHUNKS; ++it) {
        const int cur = it & 1, nxt = cur ^ 1;
        if (it + 1 < NCHUNKS) stageKV((it + 1) * 32, nxt);

        // ---- preload all 4 K^T fragments, then the 4 S WMMAs ----
        Frag kt[2][2];   // [c][hd chunk]
#pragma unroll
        for (int c = 0; c < 2; ++c) {
#pragma unroll
            for (int kc = 0; kc < 2; ++kc) {
                kt[c][kc].u[0] =
                    *(const uint4*)(&Klds[cur][c * 16 + col][kc * 32 + half * 8]);
                kt[c][kc].u[1] =
                    *(const uint4*)(&Klds[cur][c * 16 + col][kc * 32 + 16 + half * 8]);
            }
        }
        v8f s[2];
#pragma unroll
        for (int c = 0; c < 2; ++c) {
            v8f t = {};
            t = __builtin_amdgcn_wmma_f32_16x16x32_bf16(false, qf[0].v, false, kt[c][0].v,
                                                        (short)0, t, false, false);
            t = __builtin_amdgcn_wmma_f32_16x16x32_bf16(false, qf[1].v, false, kt[c][1].v,
                                                        (short)0, t, false, false);
            s[c] = t;
        }
        // ---- online softmax (row stats across the 16 lanes of each half) ----
        float mnew[8];
#pragma unroll
        for (int r = 0; r < 8; ++r) {
            s[0][r] *= 0.125f;
            s[1][r] *= 0.125f;
            float mx = fmaxf(s[0][r], s[1][r]);
            mx = fmaxf(mx, swz<1>(mx)); mx = fmaxf(mx, swz<2>(mx));
            mx = fmaxf(mx, swz<4>(mx)); mx = fmaxf(mx, swz<8>(mx));
            mnew[r] = fmaxf(mrow[r], mx);
        }
#pragma unroll
        for (int r = 0; r < 8; ++r) {
            const float p0 = __expf(s[0][r] - mnew[r]);
            const float p1 = __expf(s[1][r] - mnew[r]);
            const float alpha = __expf(mrow[r] - mnew[r]);
            mrow[r] = mnew[r];
            float ps = p0 + p1;
            ps += swz<1>(ps); ps += swz<2>(ps); ps += swz<4>(ps); ps += swz<8>(ps);
            lrow[r] = lrow[r] * alpha + ps;
            o[0][r] *= alpha; o[1][r] *= alpha; o[2][r] *= alpha; o[3][r] *= alpha;
            Slds[wave][8 * half + r][col]      = f2bf(p0);   // per-wave region,
            Slds[wave][8 * half + r][16 + col] = f2bf(p1);   // same-wave in-order
        }
        // ---- P fragment + preloaded V fragments + 4 WMMAs of P.V ----
        Frag pf, vf[4];
        pf.u[0] = *(const uint4*)(&Slds[wave][row][half * 8]);
        pf.u[1] = *(const uint4*)(&Slds[wave][row][16 + half * 8]);
#pragma unroll
        for (int t = 0; t < 4; ++t) {
            vf[t].u[0] = *(const uint4*)(&Vlds[cur][t * 16 + col][half * 8]);
            vf[t].u[1] = *(const uint4*)(&Vlds[cur][t * 16 + col][16 + half * 8]);
        }
#pragma unroll
        for (int t = 0; t < 4; ++t) {
            o[t] = __builtin_amdgcn_wmma_f32_16x16x32_bf16(
                false, pf.v, false, vf[t].v, (short)0, o[t], false, false);
        }
        wait_async0();
        __syncthreads();
    }
    // ---- epilogue: normalize, add f32 Q residual, write f32 output ----
    const float* qres = Qf + ((size_t)bb * SEQM + mBase) * DIMD + h * HD;
    float* op = out + ((size_t)bb * SEQM + mBase) * DIMD + h * HD;
#pragma unroll
    for (int t = 0; t < 4; ++t) {
#pragma unroll
        for (int r = 0; r < 8; ++r) {
            const int mr = 8 * half + r;
            const int oc = t * 16 + col;
            const float inv = 1.0f / lrow[r];
            op[(size_t)mr * DIMD + oc] = o[t][r] * inv + qres[(size_t)mr * DIMD + oc];
        }
    }
}

// ---------------------------------------------------------------------------
extern "C" void kernel_launch(void* const* d_in, const int* in_sizes, int n_in,
                              void* d_out, int out_size, void* d_ws, size_t ws_size,
                              hipStream_t stream) {
    const float* x    = (const float*)d_in[0];
    const float* ln_w = (const float*)d_in[1];
    const float* ln_b = (const float*)d_in[2];
    const float* Wk   = (const float*)d_in[3];
    const float* Wv   = (const float*)d_in[4];
    const float* Wq   = (const float*)d_in[5];
    float* out = (float*)d_out;

    char* ws = (char*)d_ws;
    size_t off = 0;
    auto take = [&](size_t bytes) { char* p = ws + off; off += (bytes + 255) & ~(size_t)255; return p; };
    unsigned short* xn  = (unsigned short*)take((size_t)BATCH * SEQN * DIMD * 2);
    unsigned short* WtK = (unsigned short*)take((size_t)KW * DIMD * DIMD * 2);
    unsigned short* WtV = (unsigned short*)take((size_t)KW * DIMD * DIMD * 2);
    unsigned short* WtQ = (unsigned short*)take((size_t)KW * DIMD * DIMD * 2);
    unsigned short* Kbf = (unsigned short*)take((size_t)BATCH * SEQN * DIMD * 2);
    unsigned short* Vtr = (unsigned short*)take((size_t)BATCH * SEQN * DIMD * 2);
    unsigned short* Qbf = (unsigned short*)take((size_t)BATCH * SEQM * DIMD * 2);
    float*          Qf  = (float*)take((size_t)BATCH * SEQM * DIMD * 4);

    ln_kernel<<<BATCH * SEQN, 256, 0, stream>>>(x, ln_w, ln_b, xn);

    const int welems = KW * DIMD * DIMD;
    wcvt_kernel<<<welems / 256, 256, 0, stream>>>(Wk, WtK);
    wcvt_kernel<<<welems / 256, 256, 0, stream>>>(Wv, WtV);
    wcvt_kernel<<<welems / 256, 256, 0, stream>>>(Wq, WtQ);

    dim3 gKV((BATCH * SEQN) / 256, DIMD / 64);
    conv_gemm_kernel<<<gKV, 256, 0, stream>>>(xn, WtK, Kbf, nullptr, nullptr, SEQN, 1);
    conv_gemm_kernel<<<gKV, 256, 0, stream>>>(xn, WtV, nullptr, nullptr, Vtr, SEQN, 1);
    dim3 gQ((BATCH * SEQM) / 256, DIMD / 64);
    conv_gemm_kernel<<<gQ, 256, 0, stream>>>(xn, WtQ, Qbf, Qf, nullptr, SEQM, 4);

    dim3 gA(SEQM / 128, BATCH * NHEADS);
    attn_kernel<<<gA, 256, 0, stream>>>(Qbf, Kbf, Vtr, Qf, out);
}
